// GRU_a_24515673325786
// MI455X (gfx1250) — matrix operand order repaired and checked by reading, not compiled, not run on hardware
//
#include <hip/hip_runtime.h>
#include <math.h>
#include <stdint.h>

// GRU-A for MI455X (gfx1250), wave32 + WMMA f32_16x16x32_f16 + async-to-LDS.
//
// One workgroup (8 wave32) owns 16 batch rows for all 336 steps; hidden state
// lives in LDS. Per step: gates(16x384) = c(16x32)@WihT + h(16x128)@WhhT via
// 15 v_wmma_f32_16x16x32_f16 per wave (fp32 accum). The x tile for step t+1
// is double-buffered into LDS with global_load_async_to_lds_b128 (ASYNCcnt),
// drained by s_wait_asynccnt at the step boundary barrier.

typedef __attribute__((ext_vector_type(16))) _Float16 v16h;
typedef __attribute__((ext_vector_type(8)))  float    v8f;

#define N_INPUT  24
#define N_HIDDEN 128
#define BATCH    4096
#define T_STEPS  336
#define G3       (3 * N_HIDDEN)   // 384
#define BM       16               // batch rows per workgroup
#define NT       24               // N tiles (384 / 16)
#define KT       4                // K tiles for Whh (128 / 32)
#define XW       (2 * N_INPUT)    // 48 floats per x row

union FragH { v16h v; uint4 q[2]; };

// --- CDNA5 async global->LDS (ISA 15.18.3 op 98): VDST = LDS byte offset,
// --- VADDR = 64-bit global address, SADDR = off. Tracked by ASYNCcnt.
__device__ __forceinline__ void async_ld_b128(uint32_t lds_off, const void* g)
{
    asm volatile("global_load_async_to_lds_b128 %0, %1, off"
                 :: "v"(lds_off), "v"((uint64_t)(uintptr_t)g)
                 : "memory");
}
__device__ __forceinline__ void wait_asynccnt0()
{
    asm volatile("s_wait_asynccnt 0x0" ::: "memory");
}

// ---------------------------------------------------------------------------
// Prep: pack Whh (384x128) and Wih (384x24 -> K padded 32) into per-lane WMMA
// B-fragment layout (f16).  B 32x16: lane&15 = column n, k = (e&8)*2 + 8*(lane>>4) + (e&7).
// Storage: [frag][lane][e], 512 halfs (1 KB) per fragment.
// ---------------------------------------------------------------------------
__global__ void pack_weights(const float* __restrict__ Wih,
                             const float* __restrict__ Whh,
                             _Float16* __restrict__ whh_p,   // [NT*KT][32][16]
                             _Float16* __restrict__ wih_p)   // [NT][32][16]
{
    int idx = blockIdx.x * blockDim.x + threadIdx.x;
    const int nWhh = NT * KT * 32 * 16;               // 49152
    if (idx < nWhh) {
        int e    = idx & 15;
        int lane = (idx >> 4) & 31;
        int f    = idx >> 9;                          // frag = nt*KT + kt
        int kt   = f & 3;
        int nt   = f >> 2;
        int n    = nt * 16 + (lane & 15);
        int k    = kt * 32 + (e & 8) * 2 + ((lane >> 4) & 1) * 8 + (e & 7);
        whh_p[idx] = (_Float16)Whh[n * N_HIDDEN + k];
    } else {
        int j = idx - nWhh;
        if (j < NT * 32 * 16) {
            int e    = j & 15;
            int lane = (j >> 4) & 31;
            int nt   = j >> 9;
            int n    = nt * 16 + (lane & 15);
            int k    = (e & 8) * 2 + ((lane >> 4) & 1) * 8 + (e & 7);    // padded K: 24..31 -> 0
            wih_p[j] = (k < N_INPUT) ? (_Float16)Wih[n * N_INPUT + k]
                                     : (_Float16)0.0f;
        }
    }
}

// ---------------------------------------------------------------------------
// Main recurrent kernel: one block = 16 batch rows, 256 threads (8 waves).
// ---------------------------------------------------------------------------
__launch_bounds__(256)
__global__ void gru_a_kernel(const float* __restrict__ x,
                             const float* __restrict__ bih,
                             const float* __restrict__ bhh,
                             const float* __restrict__ h2o_W,
                             const float* __restrict__ h2o_b,
                             const float* __restrict__ W1_W,
                             const float* __restrict__ W1_b,
                             const float* __restrict__ W2_W,
                             const float* __restrict__ W2_b,
                             const _Float16* __restrict__ whh_p,
                             const _Float16* __restrict__ wih_p,
                             float* __restrict__ out)
{
    __shared__ float    g_sum[BM][G3];        // Gx+Gh pre-activations
    __shared__ float    g_ghn[BM][N_HIDDEN];  // raw Gh for n-gate columns
    __shared__ float    h32 [BM][N_HIDDEN];   // fp32 hidden state
    __shared__ _Float16 hbf [BM][N_HIDDEN];   // f16 hidden (WMMA A operand)
    __shared__ _Float16 cbf [BM][32];         // f16 c, K padded 24->32
    __shared__ float    a1s [BM][N_INPUT];
    __shared__ float    a2s [BM][N_INPUT];
    __shared__ __align__(16) float xbuf[2][BM][XW];  // double-buffered x tile
    __shared__ float    bihL[G3], bhhL[G3];
    __shared__ float    h2oL[N_HIDDEN];
    __shared__ float    w2L [N_INPUT * N_INPUT];
    __shared__ float    w2bL[N_INPUT], w1bL[N_INPUT];

    const int tid  = threadIdx.x;
    const int lane = tid & 31;
    const int wave = tid >> 5;
    const int hi   = (lane >> 4) & 1;   // lane half (K-split select)
    const int ln   = lane & 15;         // row (A) / column (B,C,D)
    const int b0   = blockIdx.x * BM;

    // ---- one-time init -----------------------------------------------------
    if (tid < G3)      { bihL[tid] = bih[tid]; bhhL[tid] = bhh[tid]; }
    if (tid < N_HIDDEN)  h2oL[tid] = h2o_W[tid];
    if (tid < N_INPUT) { w2bL[tid] = W2_b[tid]; w1bL[tid] = W1_b[tid]; }
    for (int i = tid; i < N_INPUT * N_INPUT; i += 256) w2L[i] = W2_W[i];
    for (int i = tid; i < BM * N_HIDDEN; i += 256) {
        h32[i >> 7][i & 127] = 0.0f;
        hbf[i >> 7][i & 127] = (_Float16)0.0f;
    }
    for (int i = tid; i < BM * N_INPUT; i += 256) {
        a1s[i / N_INPUT][i % N_INPUT] = 1.0f;
        a2s[i / N_INPUT][i % N_INPUT] = 0.0f;
    }

    // async-prime x(0) into xbuf[0] (wave 0: 6 x b128 per lane batch)
    if (wave == 0) {
        const char* xbase = (const char*)(x + ((size_t)b0 * T_STEPS + 0) * XW);
        uint32_t    lbase = (uint32_t)(uintptr_t)&xbuf[0][0][0];
#pragma unroll
        for (int itc = 0; itc < 6; ++itc) {
            int c = itc * 32 + lane;
            int r = c / 12, sub = c % 12;
            async_ld_b128(lbase + (uint32_t)(r * (XW * 4) + sub * 16),
                          xbase + (size_t)r * ((size_t)T_STEPS * XW * 4) + sub * 16);
        }
        wait_asynccnt0();
    }
    __syncthreads();

    const float h2ob = h2o_b[0];

    for (int t = 0; t < T_STEPS; ++t) {
        const float (*xt)[XW] = xbuf[t & 1];

        // ---- phase 1: c = a1*h1 + a2*h2 (f16); kick async x(t+1) ----------
        for (int i = tid; i < BM * 32; i += 256) {
            int r = i >> 5, j = i & 31;
            float c = 0.0f;
            if (j < N_INPUT)
                c = a1s[r][j] * xt[r][j] + a2s[r][j] * xt[r][j + N_INPUT];
            cbf[r][j] = (_Float16)c;
        }
        if (wave == 0 && t + 1 < T_STEPS) {
            const char* xbase =
                (const char*)(x + ((size_t)b0 * T_STEPS + (t + 1)) * XW);
            uint32_t lbase = (uint32_t)(uintptr_t)&xbuf[(t + 1) & 1][0][0];
#pragma unroll
            for (int itc = 0; itc < 6; ++itc) {
                int c = itc * 32 + lane;
                int r = c / 12, sub = c % 12;
                async_ld_b128(lbase + (uint32_t)(r * (XW * 4) + sub * 16),
                              xbase + (size_t)r * ((size_t)T_STEPS * XW * 4) + sub * 16);
            }
        }
        if (t + 2 < T_STEPS && tid < BM) {   // warm L2 for the next async tile
            __builtin_prefetch(
                &x[((size_t)(b0 + tid) * T_STEPS + (t + 2)) * XW], 0, 3);
        }
        __syncthreads();

        // ---- phase 2: WMMA  gates = c @ WihT + h @ WhhT -------------------
        {
            FragH cf;   // A fragment for c (16x32)
            cf.q[0] = *(const uint4*)&cbf[ln][hi * 8];
            cf.q[1] = *(const uint4*)&cbf[ln][16 + hi * 8];

            v8f accx[3], acch[3];
#pragma unroll
            for (int i = 0; i < 3; ++i) {
                int nt = 3 * wave + i;
                FragH wf;
                const uint4* wp = (const uint4*)&wih_p[((size_t)nt * 32 + lane) * 16];
                wf.q[0] = wp[0]; wf.q[1] = wp[1];
                v8f z = {};
                accx[i] = __builtin_amdgcn_wmma_f32_16x16x32_f16(
                    false, cf.v, false, wf.v, (short)0, z, false, false);
                v8f z2 = {};
                acch[i] = z2;
            }
#pragma unroll
            for (int j = 0; j < KT; ++j) {
                FragH af;
                af.q[0] = *(const uint4*)&hbf[ln][j * 32 + hi * 8];
                af.q[1] = *(const uint4*)&hbf[ln][j * 32 + 16 + hi * 8];
#pragma unroll
                for (int i = 0; i < 3; ++i) {
                    int nt = 3 * wave + i;
                    FragH wf;
                    const uint4* wp =
                        (const uint4*)&whh_p[((size_t)(nt * KT + j) * 32 + lane) * 16];
                    wf.q[0] = wp[0]; wf.q[1] = wp[1];
                    acch[i] = __builtin_amdgcn_wmma_f32_16x16x32_f16(
                        false, af.v, false, wf.v, (short)0, acch[i], false, false);
                }
            }
            // C/D layout: VGPR v -> row v + 8*hi, column = ln
#pragma unroll
            for (int i = 0; i < 3; ++i) {
                int nt  = 3 * wave + i;
                int col = nt * 16 + ln;
#pragma unroll
                for (int v = 0; v < 8; ++v)
                    g_sum[v + hi * 8][col] = accx[i][v] + acch[i][v];
                if (nt >= 16) {  // n-gate columns need raw Gh too
#pragma unroll
                    for (int v = 0; v < 8; ++v)
                        g_ghn[v + hi * 8][col - 2 * N_HIDDEN] = acch[i][v];
                }
            }
        }
        __syncthreads();

        // ---- phase 3: gate nonlinearities + hidden update -----------------
        for (int i = tid; i < BM * N_HIDDEN; i += 256) {
            int r = i >> 7, j = i & 127;
            float rr = 1.0f / (1.0f + expf(-(g_sum[r][j] + bihL[j] + bhhL[j])));
            float zz = 1.0f / (1.0f + expf(-(g_sum[r][N_HIDDEN + j] +
                                             bihL[N_HIDDEN + j] + bhhL[N_HIDDEN + j])));
            float ghn = g_ghn[r][j];
            float gxn = g_sum[r][2 * N_HIDDEN + j] - ghn;
            float nn  = tanhf(gxn + bihL[2 * N_HIDDEN + j] +
                              rr * (ghn + bhhL[2 * N_HIDDEN + j]));
            float h   = (1.0f - zz) * nn + zz * h32[r][j];
            h32[r][j] = h;
            hbf[r][j] = (_Float16)h;
        }
        __syncthreads();

        // ---- phase 4a: scalar output head ---------------------------------
        if (tid < BM) {
            float acc = h2ob;
#pragma unroll 8
            for (int k = 0; k < N_HIDDEN; ++k) acc += h32[tid][k] * h2oL[k];
            out[(size_t)(b0 + tid) * T_STEPS + t] = tanhf(acc);
        }
        // ---- phase 4b+4c fused: w1h kept in registers, then a1/a2 ---------
        for (int i = tid; i < BM * N_INPUT; i += 256) {
            int r = i / N_INPUT, c = i % N_INPUT;
            float acc = w1bL[c];
            const float* wrow = &W1_W[c * N_HIDDEN];   // 12 KB, L2-resident
#pragma unroll 8
            for (int k = 0; k < N_HIDDEN; ++k) acc += h32[r][k] * wrow[k];
            float d1 = 0.0f, d2 = 0.0f;
#pragma unroll
            for (int k = 0; k < N_INPUT; ++k) {
                float w = w2L[c * N_INPUT + k];
                d1 += xt[r][k] * w;
                d2 += xt[r][N_INPUT + k] * w;
            }
            float base = acc + w2bL[c];
            float e1 = expf(tanhf(base + d1));
            float e2 = expf(tanhf(base + d2));
            float den = e1 + e2;
            a1s[r][c] = e1 / den;
            a2s[r][c] = e2 / den;
        }
        // drain async x(t+1) before the step-boundary barrier publishes it
        if (wave == 0) wait_asynccnt0();
        __syncthreads();
    }
}

// ---------------------------------------------------------------------------
extern "C" void kernel_launch(void* const* d_in, const int* in_sizes, int n_in,
                              void* d_out, int out_size, void* d_ws, size_t ws_size,
                              hipStream_t stream)
{
    (void)in_sizes; (void)n_in; (void)out_size; (void)ws_size;
    const float* x    = (const float*)d_in[0];
    const float* Wih  = (const float*)d_in[1];
    const float* Whh  = (const float*)d_in[2];
    const float* bih  = (const float*)d_in[3];
    const float* bhh  = (const float*)d_in[4];
    const float* h2oW = (const float*)d_in[5];
    const float* h2ob = (const float*)d_in[6];
    const float* W1W  = (const float*)d_in[7];
    const float* W1b  = (const float*)d_in[8];
    const float* W2W  = (const float*)d_in[9];
    const float* W2b  = (const float*)d_in[10];
    float* out = (float*)d_out;

    _Float16* whh_p = (_Float16*)d_ws;                    // 96 KB
    _Float16* wih_p = whh_p + (size_t)NT * KT * 32 * 16;  // +24 KB

    const int packN = (NT * KT + NT) * 32 * 16;           // 61440
    pack_weights<<<(packN + 255) / 256, 256, 0, stream>>>(Wih, Whh, whh_p, wih_p);

    gru_a_kernel<<<BATCH / BM, 256, 0, stream>>>(
        x, bih, bhh, h2oW, h2ob, W1W, W1b, W2W, W2b, whh_p, wih_p, out);
}
